// AttentionBlock_12747462935102
// MI455X (gfx1250) — matrix operand order repaired and checked by reading, not compiled, not run on hardware
//
#include <hip/hip_runtime.h>
#include <hip/hip_bf16.h>

typedef __bf16 bf16_t;
typedef __attribute__((ext_vector_type(16))) __bf16 v16bf;
typedef __attribute__((ext_vector_type(8)))  float  v8f;

constexpr int kB   = 4;
constexpr int kL   = 1024;
constexpr int kD   = 1024;
constexpr int kD4  = 4096;
constexpr int kH   = 16;
constexpr int kDH  = 64;
constexpr int kTOK = kB * kL;          // 4096 rows

typedef const __attribute__((address_space(3))) void* lds_cptr_t;

// 32-bit LDS byte offset of a __shared__ object (generic -> AS3 -> int)
__device__ __forceinline__ unsigned lds_off(const void* p) {
    return (unsigned)(uintptr_t)(lds_cptr_t)p;
}

// CDNA5 async DMA: 16B per lane, global -> LDS, tracked by ASYNCcnt
__device__ __forceinline__ void async_load_b128(unsigned lds_addr, const void* gptr) {
    unsigned long long ga = (unsigned long long)(uintptr_t)gptr;
    asm volatile("global_load_async_to_lds_b128 %0, %1, off"
                 :: "v"(lds_addr), "v"(ga) : "memory");
}

__device__ __forceinline__ void wait_async_le0() { asm volatile("s_wait_asynccnt 0x0" ::: "memory"); }
__device__ __forceinline__ void wait_async_le2() { asm volatile("s_wait_asynccnt 0x2" ::: "memory"); }
__device__ __forceinline__ void wait_async_le4() { asm volatile("s_wait_asynccnt 0x4" ::: "memory"); }

__device__ __forceinline__ float gelu_tanh(float x) {
    const float c = 0.7978845608028654f;
    float t = tanhf(c * (x + 0.044715f * x * x * x));
    return 0.5f * x * (1.0f + t);
}

// ---------------------------------------------------------------------------
// Elementwise helpers
// ---------------------------------------------------------------------------
__global__ __launch_bounds__(256) void cvt_f32_bf16_kernel(const float* __restrict__ s,
                                                           bf16_t* __restrict__ d, int n) {
    int i = (blockIdx.x * 256 + threadIdx.x) * 4;
    if (i + 3 < n) {
        #pragma unroll
        for (int j = 0; j < 4; ++j) d[i + j] = (bf16_t)s[i + j];
    }
}

__global__ __launch_bounds__(256) void copy_f32_kernel(const float* __restrict__ s,
                                                       float* __restrict__ d, int n) {
    int i = blockIdx.x * 256 + threadIdx.x;
    if (i < n) d[i] = s[i];
}

// out += (1 + alpha) * zc + z
__global__ __launch_bounds__(256) void combine_kernel(float* __restrict__ out,
                                                      const float* __restrict__ alpha,
                                                      const float* __restrict__ zc,
                                                      const float* __restrict__ z, int n) {
    int i = blockIdx.x * 256 + threadIdx.x;
    if (i < n) out[i] += (1.0f + alpha[i]) * zc[i] + z[i];
}

// Tiled transpose + f32->bf16: WT[n][k] = (bf16)W[k][n].  Grid (N/32, K/32).
__global__ __launch_bounds__(256)
void transpose_cvt_kernel(const float* __restrict__ W, bf16_t* __restrict__ WT,
                          int K, int N) {
    __shared__ float tile[32][33];
    const int k0 = blockIdx.y * 32, n0 = blockIdx.x * 32;
    const int tx = threadIdx.x & 31, ty = threadIdx.x >> 5;
    #pragma unroll
    for (int i = 0; i < 4; ++i)
        tile[ty + 8 * i][tx] = W[(size_t)(k0 + ty + 8 * i) * N + n0 + tx];
    __syncthreads();
    #pragma unroll
    for (int i = 0; i < 4; ++i)
        WT[(size_t)(n0 + ty + 8 * i) * K + k0 + tx] = (bf16_t)tile[tx][ty + 8 * i];
}

// ---------------------------------------------------------------------------
// Fused AdaLN: z = (1 + wg*rr + bg) * LN(x) + (wb*rr + bb); writes fp32 + bf16
// ---------------------------------------------------------------------------
__global__ __launch_bounds__(256)
void adaln_kernel(const float* __restrict__ X, const float* __restrict__ RR,
                  const float* __restrict__ wg, const float* __restrict__ bg,
                  const float* __restrict__ wb, const float* __restrict__ bb,
                  const float* __restrict__ lns, const float* __restrict__ lnb,
                  float* __restrict__ Z, bf16_t* __restrict__ Zb) {
    __shared__ float red[2][8];
    const int row = blockIdx.x;
    const int tid = threadIdx.x;
    const float* x = X + (size_t)row * kD;
    const float* rr = RR + (size_t)row * kD;

    float xs[4];
    float s = 0.f, s2 = 0.f;
    #pragma unroll
    for (int i = 0; i < 4; ++i) {
        xs[i] = x[tid + i * 256];
        s += xs[i];
        s2 += xs[i] * xs[i];
    }
    #pragma unroll
    for (int off = 1; off < 32; off <<= 1) {
        s += __shfl_xor(s, off, 32);
        s2 += __shfl_xor(s2, off, 32);
    }
    if ((tid & 31) == 0) { red[0][tid >> 5] = s; red[1][tid >> 5] = s2; }
    __syncthreads();
    s = 0.f; s2 = 0.f;
    #pragma unroll
    for (int w = 0; w < 8; ++w) { s += red[0][w]; s2 += red[1][w]; }
    const float mu = s * (1.0f / kD);
    const float var = s2 * (1.0f / kD) - mu * mu;
    const float rstd = rsqrtf(var + 1e-6f);

    #pragma unroll
    for (int i = 0; i < 4; ++i) {
        const int c = tid + i * 256;
        const float ln = (xs[i] - mu) * rstd * lns[c] + lnb[c];
        const float g = fmaf(wg[c], rr[c], bg[c]);
        const float be = fmaf(wb[c], rr[c], bb[c]);
        const float z = (1.0f + g) * ln + be;
        Z[(size_t)row * kD + c] = z;
        Zb[(size_t)row * kD + c] = (bf16_t)z;
    }
}

// ---------------------------------------------------------------------------
// bf16 WMMA GEMM: C[M,N] = act(A[M,K] @ BT^T + bias [+ Cacc]),  BT is [N][K].
// 256 threads = 8 waves, 128x128 block tile, wave tile 32x64 -> 8 WMMAs/K-step.
// Both tiles staged with global_load_async_to_lds_b128 into double-buffered
// LDS; s_wait_asynccnt pipelines tile t+1's DMA under tile t's WMMAs.
// ---------------------------------------------------------------------------
__global__ __launch_bounds__(256)
void gemm_bf16_wmma(const bf16_t* __restrict__ A, const bf16_t* __restrict__ BT,
                    const float* __restrict__ bias, const float* __restrict__ Cacc,
                    float* __restrict__ Cf, bf16_t* __restrict__ Cb, bf16_t* __restrict__ CbT,
                    int M, int N, int K, int act) {
    __shared__ bf16_t As[2][128][40];   // [buf][m][k]  row stride 80B
    __shared__ bf16_t Bs[2][128][40];   // [buf][n][k]

    const int tid = threadIdx.x;
    const int lane = tid & 31;
    const int wid = tid >> 5;
    const int lg = lane & 15;
    const int hig = lane >> 4;
    const int m0 = blockIdx.y * 128;
    const int n0 = blockIdx.x * 128;
    const int mrow = (wid >> 1) * 32;   // wave rows (2 M-subtiles of 16)
    const int ncol = (wid & 1) * 64;    // wave cols (4 N-subtiles of 16)

    const int lr = tid >> 2;            // loader row (both tiles 128x32)
    const int lc = (tid & 3) * 8;       // loader col group

    v8f acc[2][4];
    #pragma unroll
    for (int i = 0; i < 2; ++i)
        #pragma unroll
        for (int j = 0; j < 4; ++j) acc[i][j] = (v8f){};

    auto issue = [&](int k0, int buf) {
        async_load_b128(lds_off(&As[buf][lr][lc]),      A  + (size_t)(m0 + lr) * K + k0 + lc);
        async_load_b128(lds_off(&As[buf][64 + lr][lc]), A  + (size_t)(m0 + 64 + lr) * K + k0 + lc);
        async_load_b128(lds_off(&Bs[buf][lr][lc]),      BT + (size_t)(n0 + lr) * K + k0 + lc);
        async_load_b128(lds_off(&Bs[buf][64 + lr][lc]), BT + (size_t)(n0 + 64 + lr) * K + k0 + lc);
    };

    const int nk = K >> 5;
    issue(0, 0);
    for (int t = 0; t < nk; ++t) {
        const int cur = t & 1;
        if (t + 1 < nk) { issue((t + 1) << 5, cur ^ 1); wait_async_le4(); }
        else            { wait_async_le0(); }
        __syncthreads();

        const int akb = hig ? 16 : 0;
        v16bf a0, a1;
        #pragma unroll
        for (int i = 0; i < 8; ++i) {
            a0[i]     = As[cur][mrow + lg][akb + i];
            a0[8 + i] = As[cur][mrow + lg][akb + 8 + i];
            a1[i]     = As[cur][mrow + 16 + lg][akb + i];
            a1[8 + i] = As[cur][mrow + 16 + lg][akb + 8 + i];
        }
        const int kb2 = hig ? 16 : 0;
        #pragma unroll
        for (int j = 0; j < 4; ++j) {
            v16bf b;
            #pragma unroll
            for (int i = 0; i < 16; ++i) b[i] = Bs[cur][ncol + j * 16 + lg][kb2 + i];
            acc[0][j] = __builtin_amdgcn_wmma_f32_16x16x32_bf16(false, a0, false, b, (short)0, acc[0][j], false, false);
            acc[1][j] = __builtin_amdgcn_wmma_f32_16x16x32_bf16(false, a1, false, b, (short)0, acc[1][j], false, false);
        }
        __syncthreads();
    }

    #pragma unroll
    for (int mi = 0; mi < 2; ++mi) {
        #pragma unroll
        for (int j = 0; j < 4; ++j) {
            const int n = n0 + ncol + j * 16 + lg;
            const float bn = bias ? bias[n] : 0.0f;
            #pragma unroll
            for (int r = 0; r < 8; ++r) {
                const int m = m0 + mrow + mi * 16 + r + hig * 8;
                float v = acc[mi][j][r] + bn;
                if (Cacc) v += Cacc[(size_t)m * N + n];
                if (act) v = gelu_tanh(v);
                if (Cf)  Cf[(size_t)m * N + n] = v;
                if (Cb)  Cb[(size_t)m * N + n] = (bf16_t)v;
                if (CbT) CbT[(size_t)n * M + m] = (bf16_t)v;
            }
        }
    }
}

// ---------------------------------------------------------------------------
// Causal flash attention, scale = 1/sqrt(64).  Grid: (L/128, H, B).
// 8 waves/block, 16 q-rows each.  K tile and pre-transposed V tile both staged
// via async DMA into double-buffered LDS (2-deep ASYNCcnt pipeline).  Online
// softmax with shfl reductions over the WMMA C-layout 16-lane groups; P goes
// C-layout -> per-wave LDS -> A-layout for the PV WMMAs.
// VT is [D][TOK]: VT[h*64+dh][b*L + key].
// ---------------------------------------------------------------------------
__global__ __launch_bounds__(256)
void flash_attn_wmma(const bf16_t* __restrict__ Q, const bf16_t* __restrict__ Kc,
                     const bf16_t* __restrict__ VT, bf16_t* __restrict__ Yb) {
    __shared__ bf16_t Ks[2][32][72];    // [buf][key][dh]  row stride 144B
    __shared__ bf16_t VsT[2][64][40];   // [buf][dh][key]  row stride 80B
    __shared__ bf16_t Ps[8][16][40];    // per-wave P staging [qrow][key]

    const int tid = threadIdx.x;
    const int lane = tid & 31;
    const int wid = tid >> 5;
    const int lg = lane & 15;
    const int hig = lane >> 4;
    const int b = blockIdx.z, h = blockIdx.y, qblk = blockIdx.x;
    const int qbase = qblk * 128 + wid * 16;
    const size_t baseBH = ((size_t)b * kL) * kD + h * kDH;

    v16bf q0, q1;
    {
        const bf16_t* qp = Q + baseBH + (size_t)(qbase + lg) * kD;
        const int akb = hig ? 16 : 0;
        #pragma unroll
        for (int i = 0; i < 8; ++i) {
            q0[i] = qp[akb + i];
            q0[8 + i] = qp[akb + 8 + i];
            q1[i] = qp[32 + akb + i];
            q1[8 + i] = qp[32 + akb + 8 + i];
        }
    }

    v8f o0 = {}, o1 = {}, o2 = {}, o3 = {};
    float mrun[8], lrun[8];
    #pragma unroll
    for (int r = 0; r < 8; ++r) { mrun[r] = -1e30f; lrun[r] = 0.0f; }

    const int nkt = qblk * 4 + 4;              // tiles needed by whole block
    const int myend = (qbase + 15) >> 5;       // last tile this wave needs
    const int kb2 = hig ? 16 : 0;
    const int kr = tid >> 3, kc = (tid & 7) * 8;   // K loader: 32 x (8x8)
    const int vr = tid >> 2, vc = (tid & 3) * 8;   // V loader: 64 x (4x8)

    auto issue_kv = [&](int kt, int buf) {
        async_load_b128(lds_off(&Ks[buf][kr][kc]),
                        Kc + baseBH + (size_t)(kt * 32 + kr) * kD + kc);
        async_load_b128(lds_off(&VsT[buf][vr][vc]),
                        VT + (size_t)(h * kDH + vr) * kTOK + b * kL + kt * 32 + vc);
    };

    issue_kv(0, 0);
    for (int kt = 0; kt < nkt; ++kt) {
        const int cur = kt & 1;
        if (kt + 1 < nkt) { issue_kv(kt + 1, cur ^ 1); wait_async_le2(); }
        else              { wait_async_le0(); }
        __syncthreads();

        if (kt <= myend) {
            // ---- S = Q K^T (keys 0..15 -> s0, 16..31 -> s1) ----
            v8f s0 = {}, s1 = {};
            v16bf t;
            #pragma unroll
            for (int i = 0; i < 16; ++i) t[i] = Ks[cur][lg][kb2 + i];
            s0 = __builtin_amdgcn_wmma_f32_16x16x32_bf16(false, q0, false, t, (short)0, s0, false, false);
            #pragma unroll
            for (int i = 0; i < 16; ++i) t[i] = Ks[cur][lg][32 + kb2 + i];
            s0 = __builtin_amdgcn_wmma_f32_16x16x32_bf16(false, q1, false, t, (short)0, s0, false, false);
            #pragma unroll
            for (int i = 0; i < 16; ++i) t[i] = Ks[cur][16 + lg][kb2 + i];
            s1 = __builtin_amdgcn_wmma_f32_16x16x32_bf16(false, q0, false, t, (short)0, s1, false, false);
            #pragma unroll
            for (int i = 0; i < 16; ++i) t[i] = Ks[cur][16 + lg][32 + kb2 + i];
            s1 = __builtin_amdgcn_wmma_f32_16x16x32_bf16(false, q1, false, t, (short)0, s1, false, false);

            // ---- online softmax over this 32-key tile ----
            const int kcol0 = kt * 32 + lg;
            float p0[8], p1[8];
            #pragma unroll
            for (int r = 0; r < 8; ++r) {
                const int qi = qbase + r + hig * 8;
                float a0 = s0[r] * 0.125f; if (kcol0 > qi) a0 = -1e30f;
                float a1 = s1[r] * 0.125f; if (kcol0 + 16 > qi) a1 = -1e30f;
                float mt = fmaxf(a0, a1);
                #pragma unroll
                for (int off = 1; off < 16; off <<= 1) mt = fmaxf(mt, __shfl_xor(mt, off, 16));
                const float mn = fmaxf(mrun[r], mt);
                const float sc = __expf(mrun[r] - mn);
                mrun[r] = mn;
                const float e0 = __expf(a0 - mn);
                const float e1 = __expf(a1 - mn);
                float ts = e0 + e1;
                #pragma unroll
                for (int off = 1; off < 16; off <<= 1) ts += __shfl_xor(ts, off, 16);
                lrun[r] = lrun[r] * sc + ts;
                o0[r] *= sc; o1[r] *= sc; o2[r] *= sc; o3[r] *= sc;
                p0[r] = e0; p1[r] = e1;
            }

            // ---- P: C-layout -> LDS -> A-layout fragment ----
            #pragma unroll
            for (int r = 0; r < 8; ++r) {
                Ps[wid][r + hig * 8][lg] = (bf16_t)p0[r];
                Ps[wid][r + hig * 8][16 + lg] = (bf16_t)p1[r];
            }
            v16bf pa;
            const int pkb = hig ? 16 : 0;
            #pragma unroll
            for (int i = 0; i < 8; ++i) {
                pa[i] = Ps[wid][lg][pkb + i];
                pa[8 + i] = Ps[wid][lg][pkb + 8 + i];
            }

            // ---- O += P V ----
            #pragma unroll
            for (int j = 0; j < 4; ++j) {
                v16bf vbf;
                #pragma unroll
                for (int i = 0; i < 16; ++i) vbf[i] = VsT[cur][j * 16 + lg][kb2 + i];
                v8f& oa = (j == 0) ? o0 : (j == 1) ? o1 : (j == 2) ? o2 : o3;
                oa = __builtin_amdgcn_wmma_f32_16x16x32_bf16(false, pa, false, vbf, (short)0, oa, false, false);
            }
        }
        __syncthreads();
    }

    #pragma unroll
    for (int r = 0; r < 8; ++r) {
        const float inv = 1.0f / lrun[r];
        const int qi = qbase + r + hig * 8;
        bf16_t* yp = Yb + baseBH + (size_t)qi * kD + lg;
        yp[0]  = (bf16_t)(o0[r] * inv);
        yp[16] = (bf16_t)(o1[r] * inv);
        yp[32] = (bf16_t)(o2[r] * inv);
        yp[48] = (bf16_t)(o3[r] * inv);
    }
}

// ---------------------------------------------------------------------------
// Orchestration
// ---------------------------------------------------------------------------
extern "C" void kernel_launch(void* const* d_in, const int* in_sizes, int n_in,
                              void* d_out, int out_size, void* d_ws, size_t ws_size,
                              hipStream_t stream) {
    (void)in_sizes; (void)n_in; (void)out_size; (void)ws_size;
    auto F = [&](int i) { return (const float*)d_in[i]; };
    const float* sa = F(0);
    const float* rr = F(1);
    float* out = (float*)d_out;

    char* p = (char*)d_ws;
    auto alloc = [&](size_t bytes) {
        char* r = p;
        p += (bytes + 255) & ~(size_t)255;
        return r;
    };
    const size_t ND = (size_t)kTOK * kD;
    float*  z      = (float*)alloc(ND * 4);
    float*  zc     = (float*)alloc(ND * 4);
    float*  alphaB = (float*)alloc(ND * 4);
    bf16_t* zb  = (bf16_t*)alloc(ND * 2);
    bf16_t* rrb = (bf16_t*)alloc(ND * 2);
    bf16_t* qbf = (bf16_t*)alloc(ND * 2);
    bf16_t* kbf = (bf16_t*)alloc(ND * 2);
    bf16_t* vT  = (bf16_t*)alloc(ND * 2);   // transposed V: [D][TOK]
    bf16_t* ybf = (bf16_t*)alloc(ND * 2);
    bf16_t* zcb = (bf16_t*)alloc(ND * 2);
    bf16_t* hb  = (bf16_t*)alloc((size_t)kTOK * kD4 * 2);
    bf16_t* wT[8];                            // attn weights, transposed [N][K]
    for (int i = 0; i < 8; ++i) wT[i] = (bf16_t*)alloc((size_t)kD * kD * 2);
    bf16_t* walT0 = (bf16_t*)alloc((size_t)kD * kD * 2);
    bf16_t* walT1 = (bf16_t*)alloc((size_t)kD * kD * 2);
    bf16_t* w1T = (bf16_t*)alloc((size_t)kD * kD4 * 2);   // [4096][1024]
    bf16_t* w2T = (bf16_t*)alloc((size_t)kD4 * kD * 2);   // [1024][4096]

    dim3 blk(256);
    auto tr = [&](const float* W, bf16_t* WT, int K, int N) {
        transpose_cvt_kernel<<<dim3(N / 32, K / 32), blk, 0, stream>>>(W, WT, K, N);
    };
    auto gemm = [&](const bf16_t* A, const bf16_t* BT, const float* bias,
                    const float* Cacc, float* Cf, bf16_t* Cb, bf16_t* CbT,
                    int M, int N, int K, int act) {
        gemm_bf16_wmma<<<dim3(N / 128, M / 128), blk, 0, stream>>>(A, BT, bias, Cacc, Cf, Cb, CbT, M, N, K, act);
    };
    auto adaln = [&](const float* X, int pbase, float* Z, bf16_t* Zb) {
        adaln_kernel<<<dim3(kTOK), blk, 0, stream>>>(X, rr, F(pbase), F(pbase + 1), F(pbase + 2),
                                                     F(pbase + 3), F(pbase + 4), F(pbase + 5), Z, Zb);
    };
    auto flash = [&](const bf16_t* Q, const bf16_t* K, const bf16_t* V, bf16_t* Y) {
        flash_attn_wmma<<<dim3(kL / 128, kH, kB), blk, 0, stream>>>(Q, K, V, Y);
    };

    // once-per-launch weight transposition + conversions
    cvt_f32_bf16_kernel<<<dim3((unsigned)(ND / 1024)), blk, 0, stream>>>(rr, rrb, (int)ND);
    const int widx[8] = {20, 22, 24, 26, 28, 30, 32, 34};
    for (int i = 0; i < 8; ++i) tr(F(widx[i]), wT[i], kD, kD);
    tr(F(36), walT0, kD, kD);                       // top half of alpha W
    tr(F(36) + (size_t)kD * kD, walT1, kD, kD);     // bottom half
    tr(F(38), w1T, kD, kD4);
    tr(F(40), w2T, kD4, kD);

    // running residual lives in d_out
    copy_f32_kernel<<<dim3((unsigned)(ND / 256)), blk, 0, stream>>>(sa, out, (int)ND);

    // ---- self-attention branch ----
    adaln(sa, 2, z, zb);
    gemm(zb, wT[0], F(21), nullptr, nullptr, qbf, nullptr, kTOK, kD, kD, 0);
    gemm(zb, wT[1], F(23), nullptr, nullptr, kbf, nullptr, kTOK, kD, kD, 0);
    gemm(zb, wT[2], F(25), nullptr, nullptr, nullptr, vT, kTOK, kD, kD, 0);
    flash(qbf, kbf, vT, ybf);
    gemm(ybf, wT[3], F(27), out, out, nullptr, nullptr, kTOK, kD, kD, 0);   // sa += o-proj

    // ---- gated cross-attention branch (k = v = rr) ----
    adaln(out, 8, z, zb);
    gemm(zb,  wT[4], F(29), nullptr, nullptr, qbf, nullptr, kTOK, kD, kD, 0);
    gemm(rrb, wT[5], F(31), nullptr, nullptr, kbf, nullptr, kTOK, kD, kD, 0);
    gemm(rrb, wT[6], F(33), nullptr, nullptr, nullptr, vT, kTOK, kD, kD, 0);
    flash(qbf, kbf, vT, ybf);
    gemm(ybf, wT[7], F(35), nullptr, zc, zcb, nullptr, kTOK, kD, kD, 0);    // zc
    // alpha = [zc, z] @ Wa + b  ==  zc@Wa_top + z@Wa_bot + b
    gemm(zcb, walT0, F(37), nullptr, alphaB, nullptr, nullptr, kTOK, kD, kD, 0);
    gemm(zb, walT1, nullptr, alphaB, alphaB, nullptr, nullptr, kTOK, kD, kD, 0);
    combine_kernel<<<dim3((unsigned)(ND / 256)), blk, 0, stream>>>(out, alphaB, zc, z, (int)ND);

    // ---- MLP branch ----
    adaln(out, 14, z, zb);
    gemm(zb, w1T, F(39), nullptr, nullptr, hb, nullptr, kTOK, kD4, kD, 1);  // gelu
    gemm(hb, w2T, F(41), out, out, nullptr, nullptr, kTOK, kD, kD4, 0);     // sa += mlp
}
